// GCNNet_66383014527707
// MI455X (gfx1250) — compile-verified, matrix operand order checked by reading, not run on hardware
//
#include <hip/hip_runtime.h>
#include <hip/hip_fp16.h>

typedef __attribute__((ext_vector_type(16))) _Float16 v16h;
typedef __attribute__((ext_vector_type(8)))  _Float16 v8h;
typedef __attribute__((ext_vector_type(8)))  float    v8f;

// ---------------------------------------------------------------- utilities
__global__ void zero_f32(float* __restrict__ p, long long count) {
  long long i = (long long)blockIdx.x * blockDim.x + threadIdx.x;
  if (i < count) p[i] = 0.0f;
}

__global__ void edge_deg(const long long* __restrict__ ei, float* __restrict__ deg, int nE) {
  int e = blockIdx.x * blockDim.x + threadIdx.x;
  if (e < nE) {
    int d = (int)ei[(long long)nE + e];
    atomicAdd(&deg[d], 1.0f);
  }
}

__global__ void deg_to_dinv(float* __restrict__ deg, int n) {
  int i = blockIdx.x * blockDim.x + threadIdx.x;
  if (i < n) deg[i] = rsqrtf(deg[i] + 1.0f);
}

// ---------------------------------------------------------------- WMMA GEMM
// H[n x 32] = X[n x K] @ W[K x 32]; f16 WMMA, f32 accumulate.
// W is pre-permuted into per-lane contiguous B fragments in LDS so the hot
// loop issues ds_load_b128 pairs (no u16 gathers). One wave -> 16x32 tile.
template <int K, typename AT>
__global__ void gemm_wmma(const AT* __restrict__ X, const float* __restrict__ W,
                          float* __restrict__ H, int n) {
  constexpr int KT = K / 32;  // k-tiles of 32
  __shared__ __align__(32) _Float16 sW[KT * 2 * 32 * 16];

  // Stage W in B-fragment order: [(ktile,ctile)][lane][i] contiguous.
  for (int idx = threadIdx.x; idx < KT * 2 * 32 * 16; idx += blockDim.x) {
    int i     = idx & 15;
    int lane  = (idx >> 4) & 31;
    int ctile = (idx >> 9) & 1;
    int ktile = idx >> 10;
    int rl = lane & 15;
    int kh = lane >> 4;
    int k  = ktile * 32 + kh * 8 + (i < 8 ? i : 8 + i);  // i>=8 -> kb+16+(i-8)
    int col = rl + ctile * 16;
    sW[idx] = (_Float16)W[k * 32 + col];
  }
  __syncthreads();

  int wave = blockIdx.x * (blockDim.x >> 5) + (threadIdx.x >> 5);
  int lane = threadIdx.x & 31;
  int row0 = wave << 4;
  if (row0 >= n) return;              // wave-uniform: EXEC stays all-ones

  int rl = lane & 15;
  int kh = lane >> 4;
  int row = row0 + rl;
  int rowc = row < n ? row : (n - 1); // clamp tail loads; tail stores masked
  const AT* xrow = X + (long long)rowc * K;

  v8f acc0 = {};
  v8f acc1 = {};
#pragma unroll
  for (int ktile = 0; ktile < KT; ++ktile) {
    int kb = ktile * 32 + kh * 8;
    v16h a;
    if constexpr (__is_same(AT, _Float16)) {
      v8h lo = *(const v8h*)(xrow + kb);       // global_load_b128
      v8h hi = *(const v8h*)(xrow + kb + 16);  // global_load_b128
      a = __builtin_shufflevector(lo, hi, 0, 1, 2, 3, 4, 5, 6, 7,
                                  8, 9, 10, 11, 12, 13, 14, 15);
    } else {
#pragma unroll
      for (int i = 0; i < 8; ++i) a[i] = (_Float16)xrow[kb + i];
#pragma unroll
      for (int i = 0; i < 8; ++i) a[8 + i] = (_Float16)xrow[kb + 16 + i];
    }
    v16h b0 = *(const v16h*)(sW + ((ktile * 2 + 0) * 32 + lane) * 16);
    v16h b1 = *(const v16h*)(sW + ((ktile * 2 + 1) * 32 + lane) * 16);
    acc0 = __builtin_amdgcn_wmma_f32_16x16x32_f16(false, a, false, b0, (short)0,
                                                  acc0, false, false);
    acc1 = __builtin_amdgcn_wmma_f32_16x16x32_f16(false, a, false, b1, (short)0,
                                                  acc1, false, false);
  }
  // D layout: VGPR j -> row row0 + j + 8*kh, column rl / rl+16
#pragma unroll
  for (int j = 0; j < 8; ++j) {
    int r = row0 + j + kh * 8;
    if (r < n) {
      H[(long long)r * 32 + rl]      = acc0[j];
      H[(long long)r * 32 + rl + 16] = acc1[j];
    }
  }
}

// ---------------------------------------------------------------- edge scatter
// One wave per edge: 32 lanes = 32 feature columns. Coalesced 128B gather of
// h[src], contiguous float atomics into agg[dst] (resolved in L2).
__global__ void edge_agg(const float* __restrict__ h, const long long* __restrict__ ei,
                         const float* __restrict__ dinv, float* __restrict__ agg, int nE) {
  long long t = (long long)blockIdx.x * blockDim.x + threadIdx.x;
  int e = (int)(t >> 5);
  int c = (int)(t & 31);
  if (e >= nE) return;
  int s = (int)ei[e];
  int d = (int)ei[(long long)nE + e];
  float nm = dinv[s] * dinv[d];
  atomicAdd(&agg[(long long)d * 32 + c], h[(long long)s * 32 + c] * nm);
}

// ---------------------------------------------------------------- layer finish
// agg += h*dinv^2 + bias (self-loop), accumulate per-column sum/sumsq for BN.
__global__ void finish_bn_stats(float* __restrict__ agg, const float* __restrict__ h,
                                const float* __restrict__ dinv, const float* __restrict__ b,
                                float* __restrict__ colsum, float* __restrict__ colsq,
                                int n) {
  long long idx = (long long)blockIdx.x * blockDim.x + threadIdx.x;
  int row = (int)(idx >> 5);
  int c = (int)(idx & 31);
  float v = 0.0f;
  if (row < n) {
    float di = dinv[row];
    v = agg[idx] + h[idx] * di * di + b[c];
    agg[idx] = v;
  }
  __shared__ float ssum[256];
  __shared__ float ssq[256];
  ssum[threadIdx.x] = v;
  ssq[threadIdx.x] = v * v;
  __syncthreads();
  for (int off = 128; off >= 32; off >>= 1) {
    if (threadIdx.x < off) {
      ssum[threadIdx.x] += ssum[threadIdx.x + off];
      ssq[threadIdx.x] += ssq[threadIdx.x + off];
    }
    __syncthreads();
  }
  if (threadIdx.x < 32) {
    atomicAdd(&colsum[c], ssum[threadIdx.x]);
    atomicAdd(&colsq[c], ssq[threadIdx.x]);
  }
}

// BN + ReLU, emitting f16 activations for the next layer's WMMA A operand.
__global__ void bn_apply_f16(const float* __restrict__ agg, const float* __restrict__ colsum,
                             const float* __restrict__ colsq, const float* __restrict__ g,
                             const float* __restrict__ be, _Float16* __restrict__ out, int n) {
  long long idx = (long long)blockIdx.x * blockDim.x + threadIdx.x;
  if (idx >= (long long)n * 32) return;
  int c = (int)(idx & 31);
  float invn = 1.0f / (float)n;
  float mean = colsum[c] * invn;
  float var = colsq[c] * invn - mean * mean;
  float sc = rsqrtf(var + 1e-5f) * g[c];
  float v = (agg[idx] - mean) * sc + be[c];
  out[idx] = (_Float16)(v > 0.0f ? v : 0.0f);
}

// final layer: no BN, just add self-loop term + bias in place
__global__ void finish_final(float* __restrict__ agg, const float* __restrict__ h,
                             const float* __restrict__ dinv, const float* __restrict__ b,
                             int n) {
  long long idx = (long long)blockIdx.x * blockDim.x + threadIdx.x;
  if (idx >= (long long)n * 32) return;
  int row = (int)(idx >> 5);
  int c = (int)(idx & 31);
  float di = dinv[row];
  agg[idx] += h[idx] * di * di + b[c];
}

// ---------------------------------------------------------------- global pool
__global__ void pool_acc(const float* __restrict__ h, const long long* __restrict__ batch,
                         float* __restrict__ psum, float* __restrict__ pcnt, int n) {
  long long idx = (long long)blockIdx.x * blockDim.x + threadIdx.x;
  if (idx >= (long long)n * 32) return;
  int row = (int)(idx >> 5);
  int c = (int)(idx & 31);
  int gph = (int)batch[row];
  atomicAdd(&psum[gph * 32 + c], h[idx]);
  if (c == 0) atomicAdd(&pcnt[gph], 1.0f);
}

__global__ void pool_div(const float* __restrict__ psum, const float* __restrict__ pcnt,
                         float* __restrict__ out, int total) {
  int i = blockIdx.x * blockDim.x + threadIdx.x;
  if (i < total) {
    float cn = pcnt[i >> 5];
    out[i] = psum[i] / (cn > 1.0f ? cn : 1.0f);
  }
}

// ---------------------------------------------------------------- launcher
static inline int cdiv_i(long long a, long long b) { return (int)((a + b - 1) / b); }

extern "C" void kernel_launch(void* const* d_in, const int* in_sizes, int n_in,
                              void* d_out, int out_size, void* d_ws, size_t ws_size,
                              hipStream_t stream) {
  const float* x = (const float*)d_in[0];                 // [N,128] f32
  const long long* ei = (const long long*)d_in[1];        // [2,E] i64
  const long long* batch = (const long long*)d_in[2];     // [N] i64
  const float* W[4]  = {(const float*)d_in[3], (const float*)d_in[7],
                        (const float*)d_in[11], (const float*)d_in[15]};
  const float* bia[4] = {(const float*)d_in[4], (const float*)d_in[8],
                         (const float*)d_in[12], (const float*)d_in[16]};
  const float* gam[3] = {(const float*)d_in[5], (const float*)d_in[9],
                         (const float*)d_in[13]};
  const float* bet[3] = {(const float*)d_in[6], (const float*)d_in[10],
                         (const float*)d_in[14]};
  float* out = (float*)d_out;

  const int N = in_sizes[2];
  const int E = in_sizes[1] / 2;

  // workspace layout (floats)
  float* hbuf = (float*)d_ws;                        // N*32 f32
  float* agg  = hbuf + (long long)N * 32;            // N*32 f32
  float* abuf = agg  + (long long)N * 32;            // N*32 region (used as f16)
  float* dinv = abuf + (long long)N * 32;            // N
  float* colsum = dinv + N;                          // 32
  float* colsq  = colsum + 32;                       // 32
  float* psum = colsq + 32;                          // 64*32
  float* pcnt = psum + 64 * 32;                      // 64
  _Float16* abuf_h = (_Float16*)abuf;

  const int TB = 256;
  const long long NE32 = (long long)N * 32;

  // degrees -> dinv
  zero_f32<<<cdiv_i(N, TB), TB, 0, stream>>>(dinv, N);
  edge_deg<<<cdiv_i(E, TB), TB, 0, stream>>>(ei, dinv, E);
  deg_to_dinv<<<cdiv_i(N, TB), TB, 0, stream>>>(dinv, N);

  int waves = (N + 15) / 16;
  int gemm_blocks = (waves + 7) / 8;  // 8 waves / 256-thread block

  for (int l = 0; l < 4; ++l) {
    if (l == 0)
      gemm_wmma<128, float><<<gemm_blocks, 256, 0, stream>>>(x, W[l], hbuf, N);
    else
      gemm_wmma<32, _Float16><<<gemm_blocks, 256, 0, stream>>>(abuf_h, W[l], hbuf, N);

    zero_f32<<<cdiv_i(NE32, TB), TB, 0, stream>>>(agg, NE32);
    edge_agg<<<cdiv_i((long long)E * 32, TB), TB, 0, stream>>>(hbuf, ei, dinv, agg, E);

    if (l < 3) {
      zero_f32<<<1, 64, 0, stream>>>(colsum, 64);  // colsum+colsq contiguous
      finish_bn_stats<<<cdiv_i(NE32, TB), TB, 0, stream>>>(agg, hbuf, dinv, bia[l],
                                                           colsum, colsq, N);
      bn_apply_f16<<<cdiv_i(NE32, TB), TB, 0, stream>>>(agg, colsum, colsq, gam[l],
                                                        bet[l], abuf_h, N);
    } else {
      finish_final<<<cdiv_i(NE32, TB), TB, 0, stream>>>(agg, hbuf, dinv, bia[l], N);
    }
  }

  // global mean pool
  zero_f32<<<cdiv_i(64 * 32 + 64, TB), TB, 0, stream>>>(psum, 64 * 32 + 64);
  pool_acc<<<cdiv_i(NE32, TB), TB, 0, stream>>>(agg, batch, psum, pcnt, N);
  pool_div<<<cdiv_i(64 * 32, TB), TB, 0, stream>>>(psum, pcnt, out, 64 * 32);
}
